// Model_simple_13511967113348
// MI455X (gfx1250) — matrix operand order repaired and checked by reading, not compiled, not run on hardware
//
#include <hip/hip_runtime.h>
#include <hip/hip_bf16.h>
#include <math.h>

#define N_NODES   15840
#define IN_CH     128
#define HID       256
#define N_EDGES   253440
#define NGRAPH    48
#define NODES_PER 330

typedef __attribute__((ext_vector_type(16))) __bf16 v16bf;
typedef __attribute__((ext_vector_type(8)))  __bf16 v8bf;
typedef __attribute__((ext_vector_type(4)))  __bf16 v4bf;
typedef __attribute__((ext_vector_type(8)))  float  v8f;
typedef __attribute__((ext_vector_type(4)))  unsigned int u32x4;
typedef __attribute__((ext_vector_type(8)))  unsigned int u32x8;

__device__ __forceinline__ float wave_reduce_sum(float v) {
#pragma unroll
  for (int off = 16; off > 0; off >>= 1)
    v += __shfl_xor(v, off, 32);
  return v;
}

// ------------- LayerNorm over 128 features, one wave per node; bf16 out ------
__global__ void ln_kernel(const float* __restrict__ x, const float* __restrict__ g,
                          const float* __restrict__ b, __bf16* __restrict__ xnb) {
  int node = blockIdx.x * 8 + (threadIdx.x >> 5);
  int lane = threadIdx.x & 31;
  const float4* xr = (const float4*)(x + node * IN_CH);
  float4 v0 = xr[lane];                               // 32 lanes * 4 = 128
  float s  = v0.x + v0.y + v0.z + v0.w;
  float s2 = v0.x*v0.x + v0.y*v0.y + v0.z*v0.z + v0.w*v0.w;
  s  = wave_reduce_sum(s);
  s2 = wave_reduce_sum(s2);
  float mu  = s * (1.0f / IN_CH);
  float var = s2 * (1.0f / IN_CH) - mu * mu;
  float inv = rsqrtf(var + 1e-5f);
  int c = lane * 4;
  v4bf o;
  o[0] = (__bf16)((v0.x - mu) * inv * g[c+0] + b[c+0]);
  o[1] = (__bf16)((v0.y - mu) * inv * g[c+1] + b[c+1]);
  o[2] = (__bf16)((v0.z - mu) * inv * g[c+2] + b[c+2]);
  o[3] = (__bf16)((v0.w - mu) * inv * g[c+3] + b[c+3]);
  ((v4bf*)(xnb + node * IN_CH))[lane] = o;
}

// ------------- one-time weight conversion f32 -> bf16 ------------------------
// Wall = [4][256][128] (q,k,v,skip packed), W1b = [256][256]
__global__ void wprep_kernel(const float* __restrict__ Wq, const float* __restrict__ Wk,
                             const float* __restrict__ Wv, const float* __restrict__ Ws,
                             const float* __restrict__ W1,
                             __bf16* __restrict__ Wall, __bf16* __restrict__ W1b) {
  int i = blockIdx.x * 256 + threadIdx.x;
  if (i < 4 * HID * IN_CH) {
    int mat = i >> 15;                                 // 32768 elems per matrix
    int off = i & 32767;
    const float* src = (mat == 0) ? Wq : (mat == 1) ? Wk : (mat == 2) ? Wv : Ws;
    Wall[i] = (__bf16)src[off];
  } else {
    int j = i - 4 * HID * IN_CH;
    if (j < HID * HID) W1b[j] = (__bf16)W1[j];
  }
}

// ------------- WMMA fragment builders from bf16 memory -----------------------
// A 16x32: elems 0..7  = K kk+half*8 .. +7 ; elems 8..15 = K kk+half*8+16 .. +7
__device__ __forceinline__ v16bf a_frag(const __bf16* __restrict__ row, int kk, int half) {
  v8bf lo = *(const v8bf*)(row + kk + half * 8);
  v8bf hi = *(const v8bf*)(row + kk + half * 8 + 16);
  return __builtin_shufflevector(lo, hi, 0,1,2,3,4,5,6,7,8,9,10,11,12,13,14,15);
}
// B 32x16: elems 0..7 = K kk+half*8 .. +7 ; elems 8..15 = K kk+half*8+16 .. +7 of W row n
__device__ __forceinline__ v16bf b_frag(const __bf16* __restrict__ row, int kk, int half) {
  v8bf lo = *(const v8bf*)(row + kk + half * 8);
  v8bf hi = *(const v8bf*)(row + kk + half * 8 + 16);
  return __builtin_shufflevector(lo, hi, 0,1,2,3,4,5,6,7,8,9,10,11,12,13,14,15);
}

// ---------- Fused q/k/v/skip projections via WMMA + TDM-staged weights -------
// grid (N/32, 16), block 256 = 8 waves: wave = rowtile(2) x matrix(4).
// TDM loads the 4 weight tiles (rows col0..col0+15, K=0..127, all 4 matrices)
// as one 3D tile into LDS; waves consume B fragments from LDS.
__global__ void __launch_bounds__(256) qkvskip_kernel(
    const __bf16* __restrict__ xnb, const __bf16* __restrict__ Wall,
    const float* __restrict__ bq, const float* __restrict__ bk,
    const float* __restrict__ bv, const float* __restrict__ bs,
    float* __restrict__ q, float* __restrict__ k,
    float* __restrict__ v, float* __restrict__ skp) {
  __shared__ __bf16 Bs[4 * 16 * IN_CH];                // 16 KB
  int lane = threadIdx.x & 31;
  int wave = threadIdx.x >> 5;
  int mat  = wave & 3;
  int rowt = wave >> 2;
  int col0 = blockIdx.y * 16;
  int row0 = blockIdx.x * 32 + rowt * 16;

  if (threadIdx.x < 32) {
    // ---- Tensor DMA descriptor (D#), ISA 8.3-8.6 ----
    unsigned lds = (unsigned)(unsigned long long)(void*)&Bs[0];
    unsigned long long ga =
        (unsigned long long)(const void*)(Wall + (unsigned long long)col0 * IN_CH);
    u32x4 g0;
    g0[0] = 1u;                                        // count=1, user mode
    g0[1] = lds;                                       // lds_addr (bytes)
    g0[2] = (unsigned)ga;                              // global_addr[31:0]
    g0[3] = (unsigned)(ga >> 32) | (2u << 30);         // global_addr[56:32], type=2
    u32x8 g1;
    g1[0] = 1u << 16;                                  // mask=0, data_size=1 (2B)
    g1[1] = (unsigned)IN_CH << 16;                     // tensor_dim0 = 128
    g1[2] = (unsigned)HID << 16;                       // tensor_dim1 = 256
    g1[3] = (unsigned)IN_CH << 16;                     // tile_dim0 = 128
    g1[4] = 16u | (4u << 16);                          // tile_dim1=16, tile_dim2=4
    g1[5] = (unsigned)IN_CH;                           // tensor_dim0_stride = 128
    g1[6] = (unsigned)(HID * IN_CH) << 16;             // tensor_dim1_stride lo16 (32768)
    g1[7] = (unsigned)((HID * IN_CH) >> 16);           // tensor_dim1_stride hi
    u32x4 g2;
    g2[0] = 4u;                                        // tensor_dim2 = 4
    g2[1] = 0u; g2[2] = 0u; g2[3] = 0u;                // no dim3 / no iterate
    u32x4 g3; g3[0] = 0u; g3[1] = 0u; g3[2] = 0u; g3[3] = 0u;
    asm volatile("tensor_load_to_lds %0, %1, %2, %3"
                 :: "s"(g0), "s"(g1), "s"(g2), "s"(g3) : "memory");
    __builtin_amdgcn_s_wait_tensorcnt(0);
  }
  __syncthreads();

  int m = lane & 15, half = lane >> 4, n = lane & 15;
  v8f acc = {};
  const __bf16* arow = xnb + (unsigned long long)(row0 + m) * IN_CH;
  const __bf16* brow = &Bs[(mat * 16 + n) * IN_CH];
#pragma unroll
  for (int kk = 0; kk < IN_CH; kk += 32) {
    v16bf a = a_frag(arow, kk, half);
    v16bf b = b_frag(brow, kk, half);
    acc = __builtin_amdgcn_wmma_f32_16x16x32_bf16(false, a, false, b, (short)0, acc, false, false);
  }
  const float* bptr = (mat == 0) ? bq : (mat == 1) ? bk : (mat == 2) ? bv : bs;
  float*       optr = (mat == 0) ? q  : (mat == 1) ? k  : (mat == 2) ? v  : skp;
  float bi = bptr[col0 + n];
#pragma unroll
  for (int r = 0; r < 8; ++r) {
    int mm = r + 8 * half;
    optr[(row0 + mm) * HID + col0 + n] = acc[r] + bi;
  }
}

// ---------------- edge helpers ----------------
__device__ __forceinline__ float edge_proj(const float* __restrict__ We, int c,
                                           const float* __restrict__ ev) {
  const float* w = We + c * 5;
  return w[0]*ev[0] + w[1]*ev[1] + w[2]*ev[2] + w[3]*ev[3] + w[4]*ev[4];
}
__device__ __forceinline__ unsigned ord_of(float f) {
  unsigned u = __float_as_uint(f);
  return (u & 0x80000000u) ? ~u : (u | 0x80000000u);
}
__device__ __forceinline__ float ord_dec(unsigned u) {
  return (u & 0x80000000u) ? __uint_as_float(u & 0x7FFFFFFFu) : __uint_as_float(~u);
}

__global__ void init_kernel(unsigned* __restrict__ mord, float* __restrict__ z) {
  int i = blockIdx.x * 256 + threadIdx.x;
  if (i < N_NODES) { mord[i] = 0x007FFFFFu /* ord(-inf) */; z[i] = 0.0f; }
}

// alpha_e = dot(q[dst], k[src]+e)/16 ; segment max via ordered-uint atomicMax
__global__ void edge_alpha_kernel(const int* __restrict__ ei, const float* __restrict__ ea,
                                  const float* __restrict__ We,
                                  const float* __restrict__ q, const float* __restrict__ k,
                                  float* __restrict__ alpha, unsigned* __restrict__ mord) {
  int e    = blockIdx.x * 8 + (threadIdx.x >> 5);
  int lane = threadIdx.x & 31;
  int src = ei[e], dst = ei[N_EDGES + e];
  __builtin_prefetch(ea + (e + 8) * 5, 0, 0);          // global_prefetch_b8
  float ev[5];
#pragma unroll
  for (int d = 0; d < 5; ++d) ev[d] = ea[e * 5 + d];
  const float4* qd = (const float4*)(q + (long)dst * HID);
  const float4* ks = (const float4*)(k + (long)src * HID);
  float s = 0.0f;
#pragma unroll
  for (int j = 0; j < 2; ++j) {
    float4 qv = qd[lane * 2 + j];
    float4 kv = ks[lane * 2 + j];
    int c = lane * 8 + j * 4;
    s += qv.x * (kv.x + edge_proj(We, c + 0, ev));
    s += qv.y * (kv.y + edge_proj(We, c + 1, ev));
    s += qv.z * (kv.z + edge_proj(We, c + 2, ev));
    s += qv.w * (kv.w + edge_proj(We, c + 3, ev));
  }
  s = wave_reduce_sum(s);
  if (lane == 0) {
    float al = s * 0.0625f;                            // 1/sqrt(256)
    alpha[e] = al;
    atomicMax(&mord[dst], ord_of(al));
  }
}

__global__ void edge_exp_kernel(const int* __restrict__ ei, float* __restrict__ alpha,
                                const unsigned* __restrict__ mord, float* __restrict__ z) {
  int e = blockIdx.x * 256 + threadIdx.x;
  if (e >= N_EDGES) return;
  int dst = ei[N_EDGES + e];
  float a = __expf(alpha[e] - ord_dec(mord[dst]));
  alpha[e] = a;
  atomicAdd(&z[dst], a);
}

// out[dst] += (a/z[dst]) * (v[src] + e)
__global__ void edge_agg_kernel(const int* __restrict__ ei, const float* __restrict__ ea,
                                const float* __restrict__ We, const float* __restrict__ v,
                                const float* __restrict__ alpha, const float* __restrict__ z,
                                float* __restrict__ outb) {
  int e    = blockIdx.x * 8 + (threadIdx.x >> 5);
  int lane = threadIdx.x & 31;
  int src = ei[e], dst = ei[N_EDGES + e];
  float coef = alpha[e] / (z[dst] + 1e-16f);
  float ev[5];
#pragma unroll
  for (int d = 0; d < 5; ++d) ev[d] = ea[e * 5 + d];
  const float4* vs = (const float4*)(v + (long)src * HID);
  float* od = outb + (long)dst * HID;
#pragma unroll
  for (int j = 0; j < 2; ++j) {
    float4 vv = vs[lane * 2 + j];
    int c = lane * 8 + j * 4;
    atomicAdd(od + c + 0, coef * (vv.x + edge_proj(We, c + 0, ev)));
    atomicAdd(od + c + 1, coef * (vv.y + edge_proj(We, c + 1, ev)));
    atomicAdd(od + c + 2, coef * (vv.z + edge_proj(We, c + 2, ev)));
    atomicAdd(od + c + 3, coef * (vv.w + edge_proj(We, c + 3, ev)));
  }
}

// ---------------- BatchNorm (training stats) + ReLU ----------------
__global__ void bn_stats_kernel(const float* __restrict__ outb,
                                float* __restrict__ bm, float* __restrict__ istd) {
  int c = blockIdx.x;
  float s = 0.0f, s2 = 0.0f;
  for (int i = threadIdx.x; i < N_NODES; i += 256) {
    float v = outb[(long)i * HID + c];
    s += v; s2 += v * v;
  }
  __shared__ float sh[256], sh2[256];
  sh[threadIdx.x] = s; sh2[threadIdx.x] = s2;
  __syncthreads();
  for (int o = 128; o > 0; o >>= 1) {
    if (threadIdx.x < o) { sh[threadIdx.x] += sh[threadIdx.x + o]; sh2[threadIdx.x] += sh2[threadIdx.x + o]; }
    __syncthreads();
  }
  if (threadIdx.x == 0) {
    float mu  = sh[0] * (1.0f / N_NODES);
    float var = sh2[0] * (1.0f / N_NODES) - mu * mu;
    bm[c] = mu;
    istd[c] = rsqrtf(var + 1e-5f);
  }
}

__global__ void bn_apply_kernel(float* __restrict__ outb, const float* __restrict__ bm,
                                const float* __restrict__ istd, const float* __restrict__ g,
                                const float* __restrict__ b) {
  int i = blockIdx.x * 256 + threadIdx.x;
  int c = i & (HID - 1);
  float v = (outb[i] - bm[c]) * istd[c] * g[c] + b[c];
  outb[i] = v > 0.0f ? v : 0.0f;
}

// ------------- pooling 330 -> 18: max over 18 consecutive nodes; bf16 out ----
__global__ void pool_kernel(const float* __restrict__ outb, __bf16* __restrict__ hpb) {
  int bt = blockIdx.x;                  // 0..48*18-1
  int c  = threadIdx.x;                 // 256
  int b = bt / 18, t = bt % 18;
  const float* base = outb + ((long)(b * NODES_PER + t * 18)) * HID + c;
  float m = -3.402823466e+38f;
#pragma unroll
  for (int p = 0; p < 18; ++p) m = fmaxf(m, base[p * HID]);
  hpb[(long)bt * HID + c] = (__bf16)m;
}

// ---------------- MLP: h1 = relu(hp @ W1^T + b1), 864x256x256 via WMMA -------
__global__ void mlp_wmma_kernel(const __bf16* __restrict__ hpb, const __bf16* __restrict__ W1b,
                                const float* __restrict__ b1, float* __restrict__ h1) {
  int lane = threadIdx.x & 31;
  int wave = threadIdx.x >> 5;
  int row0 = blockIdx.x * 16;
  int col0 = (blockIdx.y * 4 + wave) * 16;
  int m = lane & 15, half = lane >> 4, n = lane & 15;
  v8f acc = {};
  const __bf16* arow = hpb + (row0 + m) * HID;
  const __bf16* brow = W1b + (col0 + n) * HID;
#pragma unroll
  for (int kk = 0; kk < HID; kk += 32) {
    v16bf a = a_frag(arow, kk, half);
    v16bf b = b_frag(brow, kk, half);
    acc = __builtin_amdgcn_wmma_f32_16x16x32_bf16(false, a, false, b, (short)0, acc, false, false);
  }
  float bi = b1[col0 + n];
#pragma unroll
  for (int r = 0; r < 8; ++r) {
    int mm = r + 8 * half;
    float vv = acc[r] + bi;
    h1[(row0 + mm) * HID + col0 + n] = vv > 0.0f ? vv : 0.0f;
  }
}

// ---------------- readout: sigmoid(mean_t(h1 @ Wr^T + br)) per graph ---------
__global__ void readout_kernel(const float* __restrict__ h1, const float* __restrict__ Wr,
                               const float* __restrict__ br, float* __restrict__ out) {
  int b = blockIdx.x;
  float s = 0.0f;
  for (int i = threadIdx.x; i < 18 * HID; i += 256) {
    int c = i & (HID - 1);
    s += h1[(long)b * 18 * HID + i] * Wr[c];
  }
  __shared__ float sh[256];
  sh[threadIdx.x] = s;
  __syncthreads();
  for (int o = 128; o > 0; o >>= 1) {
    if (threadIdx.x < o) sh[threadIdx.x] += sh[threadIdx.x + o];
    __syncthreads();
  }
  if (threadIdx.x == 0) {
    float c = sh[0] * (1.0f / 18.0f) + br[0];
    out[b] = 1.0f / (1.0f + __expf(-c));
  }
}

extern "C" void kernel_launch(void* const* d_in, const int* in_sizes, int n_in,
                              void* d_out, int out_size, void* d_ws, size_t ws_size,
                              hipStream_t stream) {
  const float* x    = (const float*)d_in[0];
  const int*   ei   = (const int*)d_in[1];
  const float* ea   = (const float*)d_in[2];
  // d_in[3] = batch (implied by node index; unused)
  const float* Wq   = (const float*)d_in[4];  const float* bq  = (const float*)d_in[5];
  const float* Wk   = (const float*)d_in[6];  const float* bk  = (const float*)d_in[7];
  const float* Wv   = (const float*)d_in[8];  const float* bv  = (const float*)d_in[9];
  const float* We   = (const float*)d_in[10];
  const float* Wsk  = (const float*)d_in[11]; const float* bsk = (const float*)d_in[12];
  const float* ln_g = (const float*)d_in[13]; const float* ln_b = (const float*)d_in[14];
  const float* bn_g = (const float*)d_in[15]; const float* bn_b = (const float*)d_in[16];
  const float* W1   = (const float*)d_in[17]; const float* b1  = (const float*)d_in[18];
  const float* Wr   = (const float*)d_in[19]; const float* br  = (const float*)d_in[20];

  // ---- workspace layout: bf16 region first (16B-aligned boundary), then f32
  __bf16* xnb  = (__bf16*)d_ws;                        // N*128
  __bf16* Wall = xnb + (size_t)N_NODES * IN_CH;        // 4*256*128
  __bf16* W1b  = Wall + 4 * HID * IN_CH;               // 256*256
  __bf16* hpb  = W1b + HID * HID;                      // 864*256
  float* q     = (float*)(hpb + (size_t)NGRAPH * 18 * HID);
  float* k     = q    + (size_t)N_NODES * HID;
  float* v     = k    + (size_t)N_NODES * HID;
  float* outb  = v    + (size_t)N_NODES * HID;         // skip + aggregation
  float* alpha = outb + (size_t)N_NODES * HID;         // E
  unsigned* mord = (unsigned*)(alpha + N_EDGES);       // N
  float* z     = (float*)(mord + N_NODES);             // N
  float* bm    = z + N_NODES;                          // 256
  float* istd  = bm + HID;                             // 256
  float* h1    = istd + HID;                           // 864*256

  ln_kernel<<<N_NODES / 8, 256, 0, stream>>>(x, ln_g, ln_b, xnb);
  wprep_kernel<<<(4 * HID * IN_CH + HID * HID) / 256, 256, 0, stream>>>(
      Wq, Wk, Wv, Wsk, W1, Wall, W1b);
  qkvskip_kernel<<<dim3(N_NODES / 32, HID / 16), 256, 0, stream>>>(
      xnb, Wall, bq, bk, bv, bsk, q, k, v, outb);
  init_kernel<<<(N_NODES + 255) / 256, 256, 0, stream>>>(mord, z);
  edge_alpha_kernel<<<N_EDGES / 8, 256, 0, stream>>>(ei, ea, We, q, k, alpha, mord);
  edge_exp_kernel<<<(N_EDGES + 255) / 256, 256, 0, stream>>>(ei, alpha, mord, z);
  edge_agg_kernel<<<N_EDGES / 8, 256, 0, stream>>>(ei, ea, We, v, alpha, z, outb);
  bn_stats_kernel<<<HID, 256, 0, stream>>>(outb, bm, istd);
  bn_apply_kernel<<<(N_NODES * HID) / 256, 256, 0, stream>>>(outb, bm, istd, bn_g, bn_b);
  pool_kernel<<<NGRAPH * 18, 256, 0, stream>>>(outb, hpb);
  mlp_wmma_kernel<<<dim3((NGRAPH * 18) / 16, 4), 128, 0, stream>>>(hpb, W1b, b1, h1);
  readout_kernel<<<NGRAPH, 256, 0, stream>>>(h1, Wr, br, (float*)d_out);
}
